// Denoiser_77841987273287
// MI455X (gfx1250) — matrix operand order repaired and checked by reading, not compiled, not run on hardware
//
#include <hip/hip_runtime.h>
#include <hip/hip_bf16.h>

typedef _Float16 h8  __attribute__((ext_vector_type(8)));
typedef _Float16 h16 __attribute__((ext_vector_type(16)));
typedef float    f8  __attribute__((ext_vector_type(8)));

#define NPTS 8192
#define BATCH 8
#define MS 1024
#define KNNK 17

// ---------------------------------------------------------------------------
// WMMA GEMM: out[pos][mo] = act( dot(A[mo][:], Bm[pos][:]) + bias[mo] )
// A: [MO][K] f16 row-major (weights), Bm: [NO][K] f16 (activations, channel-
// contiguous per position). One 16x16 output tile per wave, K-loop step 32.
// A-fragment: lane<16 holds row m0+lane, K {k0..k0+15};
//             lane>=16 holds K {k0+16..k0+31}  (contiguous, matches B).
// B-fragment: lane half selects contiguous 16 K values  (same contiguous order as A).
// D: lane = column (position), VGPR r = out-channel m0 + r + 8*(lane>=16).
// ---------------------------------------------------------------------------
template<int RELU, int F16OUT>
__global__ __launch_bounds__(256)
void gemm_wmma(const _Float16* __restrict__ A, const _Float16* __restrict__ Bm,
               const float* __restrict__ bias, _Float16* __restrict__ out16,
               float* __restrict__ out32, int MO, int K, int NO)
{
  const int lane = threadIdx.x & 31;
  const int wave = threadIdx.x >> 5;
  const int tilesM = MO >> 4;
  const int tile = blockIdx.x * 8 + wave;
  const int totTiles = tilesM * ((NO + 15) >> 4);
  if (tile >= totTiles) return;                 // wave-uniform exit (EXEC stays all-1 for WMMA)
  const int m0 = (tile % tilesM) << 4;
  const int n0 = (tile / tilesM) << 4;
  const int lh = lane & 15;
  const int hb = lane >> 4;

  const _Float16* arow = A + (size_t)(m0 + lh) * K + hb * 16;
  int bn = n0 + lh; if (bn >= NO) bn = NO - 1;  // clamp so every lane loads valid memory
  const _Float16* brow = Bm + (size_t)bn * K + hb * 16;

  f8 acc = {0.f,0.f,0.f,0.f,0.f,0.f,0.f,0.f};
  for (int k0 = 0; k0 < K; k0 += 32) {
    h8 alo = *(const h8*)(arow + k0);
    h8 ahi = *(const h8*)(arow + k0 + 8);
    h16 afrag = __builtin_shufflevector(alo, ahi,
        0,1,2,3,4,5,6,7,8,9,10,11,12,13,14,15);
    h16 bfrag = *(const h16*)(brow + k0);
    acc = __builtin_amdgcn_wmma_f32_16x16x32_f16(
        false, afrag, false, bfrag, (short)0, acc, false, false);
  }
  const int pos = n0 + lh;
  if (pos < NO) {
    const int cb = m0 + hb * 8;
    #pragma unroll
    for (int r = 0; r < 8; ++r) {
      float v = acc[r];
      if (bias) v += bias[cb + r];
      if (RELU) v = fmaxf(v, 0.0f);
      if (F16OUT) out16[(size_t)pos * MO + cb + r] = (_Float16)v;
      else        out32[(size_t)pos * MO + cb + r] = v;
    }
  }
}

// ---------------------------------------------------------------------------
// Weight prep: f32 -> f16 copies (+ transpose of k_w for the K^T·q trick)
// ---------------------------------------------------------------------------
__global__ __launch_bounds__(256)
void prep_kernel(const float* __restrict__ csw1, const float* __restrict__ csw2,
                 const float* __restrict__ qw, const float* __restrict__ kw,
                 _Float16* __restrict__ o1, _Float16* __restrict__ o2,
                 _Float16* __restrict__ o3, _Float16* __restrict__ o4T)
{
  int i = blockIdx.x * 256 + threadIdx.x;
  if (i < 256*256) o1[i] = (_Float16)csw1[i];
  if (i < 128*256) o2[i] = (_Float16)csw2[i];
  if (i < 256*256) o3[i] = (_Float16)qw[i];
  if (i < 256*256) { int rr = i >> 8, cc = i & 255; o4T[(size_t)cc*256 + rr] = (_Float16)kw[i]; }
}

// ---------------------------------------------------------------------------
// Global-feature MLP: 512 -> 256 relu -> 128, one block per batch
// ---------------------------------------------------------------------------
__global__ __launch_bounds__(256)
void gfeat_kernel(const float* __restrict__ gf, const float* __restrict__ w1,
                  const float* __restrict__ b1, const float* __restrict__ w2,
                  const float* __restrict__ b2, _Float16* __restrict__ g16)
{
  __shared__ float h[256];
  const int b = blockIdx.x, t = threadIdx.x;
  const float* gin = gf + (size_t)b * 512;
  float s = b1[t];
  for (int j = 0; j < 512; ++j) s += w1[t*512+j] * gin[j];
  h[t] = fmaxf(s, 0.f);
  __syncthreads();
  if (t < 128) {
    float s2 = b2[t];
    for (int j = 0; j < 256; ++j) s2 += w2[t*256+j] * h[j];
    g16[b*128 + t] = (_Float16)s2;
  }
}

// ---------------------------------------------------------------------------
// Per-point MLP 3 -> 64 relu -> 128 and build xcat = [f1 | g_b] as f16[p][256]
// ---------------------------------------------------------------------------
__global__ __launch_bounds__(256)
void pointmlp_kernel(const float* __restrict__ x, const float* __restrict__ w1,
                     const float* __restrict__ b1, const float* __restrict__ w2,
                     const float* __restrict__ b2, const _Float16* __restrict__ g16,
                     _Float16* __restrict__ xcat)
{
  __shared__ float sw2[128*64];
  const int t = threadIdx.x;
  for (int i = t; i < 128*64; i += 256) sw2[i] = w2[i];
  __syncthreads();
  const int p = blockIdx.x * 256 + t;          // p = b*NPTS + n  (x is [B][N][3])
  const int b = p >> 13;
  const float* xp = x + (size_t)p * 3;
  const float px = xp[0], py = xp[1], pz = xp[2];
  float h[64];
  #pragma unroll
  for (int c = 0; c < 64; ++c)
    h[c] = fmaxf(w1[c*3+0]*px + w1[c*3+1]*py + w1[c*3+2]*pz + b1[c], 0.0f);
  _Float16* op = xcat + (size_t)p * 256;
  for (int o = 0; o < 128; ++o) {
    float s = b2[o];
    #pragma unroll 8
    for (int j = 0; j < 64; ++j) s += sw2[o*64+j] * h[j];
    op[o] = (_Float16)s;
  }
  const _Float16* gb = g16 + b * 128;
  for (int o = 0; o < 128; ++o) op[128 + o] = gb[o];
}

// ---------------------------------------------------------------------------
// Farthest point sampling: one block per batch, min-dist array in LDS,
// 1023 dependent argmax steps with a 256-wide tree reduction each.
// Also emits down_x straight into d_out.
// ---------------------------------------------------------------------------
__global__ __launch_bounds__(256)
void fps_kernel(const float* __restrict__ x, int* __restrict__ fpsidx,
                float* __restrict__ downx)
{
  __shared__ float mind[NPTS];
  __shared__ float red[256];
  __shared__ int   redi[256];
  __shared__ int   sidx[MS];
  __shared__ float lastc[3];
  const int b = blockIdx.x, t = threadIdx.x;
  const float* xb = x + (size_t)b * NPTS * 3;
  for (int i = t; i < NPTS; i += 256) mind[i] = 1e10f;
  if (t == 0) sidx[0] = 0;
  int last = 0;
  __syncthreads();
  for (int s = 1; s < MS; ++s) {
    if (t == 0) { lastc[0]=xb[last*3]; lastc[1]=xb[last*3+1]; lastc[2]=xb[last*3+2]; }
    __syncthreads();
    const float lx = lastc[0], ly = lastc[1], lz = lastc[2];
    float bmax = -1.0f; int bid = 0;
    for (int i = t; i < NPTS; i += 256) {
      float dx = xb[i*3]-lx, dy = xb[i*3+1]-ly, dz = xb[i*3+2]-lz;
      float m = fminf(mind[i], dx*dx + dy*dy + dz*dz);
      mind[i] = m;
      if (m > bmax) { bmax = m; bid = i; }
    }
    red[t] = bmax; redi[t] = bid;
    __syncthreads();
    for (int st = 128; st > 0; st >>= 1) {
      if (t < st && red[t+st] > red[t]) { red[t] = red[t+st]; redi[t] = redi[t+st]; }
      __syncthreads();
    }
    last = redi[0];
    if (t == 0) sidx[s] = last;
    __syncthreads();
  }
  for (int s = t; s < MS; s += 256) {
    const int id = sidx[s];
    fpsidx[b*MS + s] = id;
    downx[((size_t)b*MS + s)*3 + 0] = xb[id*3 + 0];
    downx[((size_t)b*MS + s)*3 + 1] = xb[id*3 + 1];
    downx[((size_t)b*MS + s)*3 + 2] = xb[id*3 + 2];
  }
}

// ---------------------------------------------------------------------------
// KNN: one block per (b,m) query; distances in LDS; 17 argmin sweeps
// ---------------------------------------------------------------------------
__global__ __launch_bounds__(256)
void knn_kernel(const float* __restrict__ x, const float* __restrict__ downx,
                int* __restrict__ knnidx)
{
  __shared__ float dist[NPTS];
  __shared__ float red[256];
  __shared__ int   redi[256];
  const int bm = blockIdx.x, t = threadIdx.x;
  const int b = bm >> 10;
  const float* xb = x + (size_t)b * NPTS * 3;
  const float qx = downx[(size_t)bm*3], qy = downx[(size_t)bm*3+1], qz = downx[(size_t)bm*3+2];
  for (int i = t; i < NPTS; i += 256) {
    float dx = xb[i*3]-qx, dy = xb[i*3+1]-qy, dz = xb[i*3+2]-qz;
    dist[i] = dx*dx + dy*dy + dz*dz;
  }
  __syncthreads();
  for (int kk = 0; kk < KNNK; ++kk) {
    float bmin = 1e30f; int bid = 0;
    for (int i = t; i < NPTS; i += 256) {
      float d = dist[i];
      if (d < bmin) { bmin = d; bid = i; }
    }
    red[t] = bmin; redi[t] = bid;
    __syncthreads();
    for (int st = 128; st > 0; st >>= 1) {
      if (t < st && red[t+st] < red[t]) { red[t] = red[t+st]; redi[t] = redi[t+st]; }
      __syncthreads();
    }
    if (t == 0) { knnidx[bm*KNNK + kk] = redi[0]; dist[redi[0]] = 1e30f; }
    __syncthreads();
  }
}

// ---------------------------------------------------------------------------
// Local geometry convs (9->64->64->128, BN eval + leaky 0.01) and feat build:
// feat = [knn_f(128) | r2(128)] f16, split into featq (k=0) / featk (k=1..16)
// ---------------------------------------------------------------------------
__global__ __launch_bounds__(128)
void local_kernel(const float* __restrict__ x, const _Float16* __restrict__ fbuf,
                  const int* __restrict__ knnidx, const float* __restrict__ downx,
                  const float* __restrict__ w1, const float* __restrict__ b1,
                  const float* __restrict__ g1, const float* __restrict__ bb1,
                  const float* __restrict__ w2, const float* __restrict__ b2,
                  const float* __restrict__ g2, const float* __restrict__ bb2,
                  const float* __restrict__ w3, const float* __restrict__ b3,
                  _Float16* __restrict__ featq, _Float16* __restrict__ featk)
{
  __shared__ float r[9][KNNK];
  __shared__ float h1s[64][KNNK];
  __shared__ float h2s[64][KNNK];
  __shared__ int sidx[KNNK];
  const int bm = blockIdx.x, t = threadIdx.x;
  const int b = bm >> 10;
  const float* xb = x + (size_t)b * NPTS * 3;
  if (t < KNNK) sidx[t] = knnidx[bm*KNNK + t];
  const float cx = downx[(size_t)bm*3], cy = downx[(size_t)bm*3+1], cz = downx[(size_t)bm*3+2];
  __syncthreads();
  if (t < KNNK) {
    const int id = sidx[t];
    const float px = xb[id*3], py = xb[id*3+1], pz = xb[id*3+2];
    r[0][t]=cx; r[1][t]=cy; r[2][t]=cz;
    r[3][t]=px; r[4][t]=py; r[5][t]=pz;
    r[6][t]=cx-px; r[7][t]=cy-py; r[8][t]=cz-pz;
  }
  __syncthreads();
  const float bninv = rsqrtf(1.0f + 1e-5f);
  for (int e = t; e < 64*KNNK; e += 128) {
    const int c = e / KNNK, kk = e % KNNK;
    float s = b1[c];
    #pragma unroll
    for (int j = 0; j < 9; ++j) s += w1[c*9+j] * r[j][kk];
    s = s * (g1[c]*bninv) + bb1[c];
    h1s[c][kk] = (s > 0.f) ? s : 0.01f*s;
  }
  __syncthreads();
  for (int e = t; e < 64*KNNK; e += 128) {
    const int c = e / KNNK, kk = e % KNNK;
    float s = b2[c];
    for (int j = 0; j < 64; ++j) s += w2[c*64+j] * h1s[j][kk];
    s = s * (g2[c]*bninv) + bb2[c];
    h2s[c][kk] = (s > 0.f) ? s : 0.01f*s;
  }
  __syncthreads();
  for (int e = t; e < 256*KNNK; e += 128) {
    const int c = e / KNNK, kk = e % KNNK;
    float v;
    if (c < 128) {
      const int id = sidx[kk];
      v = (float)fbuf[((size_t)b*NPTS + id)*128 + c];
    } else {
      const int o = c - 128;
      float s = b3[o];
      for (int j = 0; j < 64; ++j) s += w3[o*64+j] * h2s[j][kk];
      v = s;
    }
    const _Float16 hv = (_Float16)v;
    if (kk == 0) featq[(size_t)bm*256 + c] = hv;
    else         featk[((size_t)bm*16 + (kk-1))*256 + c] = hv;
  }
}

// ---------------------------------------------------------------------------
// Attention tail: logits[kk] = dot(K_w^T q, featk) (q.k_b drops out of the
// softmax since it is constant over kk); softmax; weighted coord sum.
// ---------------------------------------------------------------------------
__global__ __launch_bounds__(32)
void attn_kernel(const float* __restrict__ kt, const _Float16* __restrict__ featk,
                 const int* __restrict__ knnidx, const float* __restrict__ x,
                 float* __restrict__ newx)
{
  __shared__ float lg[16];
  const int bm = blockIdx.x, t = threadIdx.x;
  const int b = bm >> 10;
  if (t < 16) {
    const float* ktb = kt + (size_t)bm * 256;
    const _Float16* fk = featk + ((size_t)bm*16 + t) * 256;
    float s = 0.f;
    for (int c = 0; c < 256; ++c) s += ktb[c] * (float)fk[c];
    lg[t] = s;
  }
  __syncthreads();
  if (t == 0) {
    float mx = lg[0];
    for (int k = 1; k < 16; ++k) mx = fmaxf(mx, lg[k]);
    float ex[16]; float sum = 0.f;
    for (int k = 0; k < 16; ++k) { ex[k] = __expf(lg[k]-mx); sum += ex[k]; }
    const float* xb = x + (size_t)b * NPTS * 3;
    float ox = 0.f, oy = 0.f, oz = 0.f;
    for (int k = 0; k < 16; ++k) {
      const int id = knnidx[bm*KNNK + 1 + k];
      const float ww = ex[k] / sum;
      ox += ww*xb[id*3]; oy += ww*xb[id*3+1]; oz += ww*xb[id*3+2];
    }
    newx[(size_t)bm*3+0] = ox; newx[(size_t)bm*3+1] = oy; newx[(size_t)bm*3+2] = oz;
  }
}

// ---------------------------------------------------------------------------
extern "C" void kernel_launch(void* const* d_in, const int* in_sizes, int n_in,
                              void* d_out, int out_size, void* d_ws, size_t ws_size,
                              hipStream_t stream) {
  (void)in_sizes; (void)n_in; (void)out_size; (void)ws_size;
  const float* x     = (const float*)d_in[0];
  const float* gfeat = (const float*)d_in[1];
  const float* c1w1  = (const float*)d_in[2];
  const float* c1b1  = (const float*)d_in[3];
  const float* c1w2  = (const float*)d_in[4];
  const float* c1b2  = (const float*)d_in[5];
  const float* cfw1  = (const float*)d_in[6];
  const float* cfb1  = (const float*)d_in[7];
  const float* cfw2  = (const float*)d_in[8];
  const float* cfb2  = (const float*)d_in[9];
  const float* csw1  = (const float*)d_in[10];
  const float* csb1  = (const float*)d_in[11];
  const float* csw2  = (const float*)d_in[12];
  const float* csb2  = (const float*)d_in[13];
  const float* s2w1  = (const float*)d_in[14];
  const float* s2b1  = (const float*)d_in[15];
  const float* s2g1  = (const float*)d_in[16];
  const float* s2bb1 = (const float*)d_in[17];
  const float* s2w2  = (const float*)d_in[18];
  const float* s2b2  = (const float*)d_in[19];
  const float* s2g2  = (const float*)d_in[20];
  const float* s2bb2 = (const float*)d_in[21];
  const float* s2w3  = (const float*)d_in[22];
  const float* s2b3  = (const float*)d_in[23];
  const float* qw    = (const float*)d_in[24];
  const float* qb    = (const float*)d_in[25];
  const float* kw    = (const float*)d_in[26];

  char* wsb = (char*)d_ws;
  size_t off = 0;
  auto take = [&](size_t bytes) { char* p = wsb + off; off = (off + bytes + 255) & ~(size_t)255; return p; };

  _Float16* csw1_16 = (_Float16*)take((size_t)256*256*2);
  _Float16* csw2_16 = (_Float16*)take((size_t)128*256*2);
  _Float16* qw16    = (_Float16*)take((size_t)256*256*2);
  _Float16* kwT16   = (_Float16*)take((size_t)256*256*2);
  _Float16* g16     = (_Float16*)take((size_t)BATCH*128*2);
  _Float16* xcat    = (_Float16*)take((size_t)BATCH*NPTS*256*2);
  _Float16* hbuf    = (_Float16*)take((size_t)BATCH*NPTS*256*2);
  _Float16* fbuf    = (_Float16*)take((size_t)BATCH*NPTS*128*2);
  int*      fpsidx  = (int*)     take((size_t)BATCH*MS*4);
  int*      knnidx  = (int*)     take((size_t)BATCH*MS*KNNK*4);
  _Float16* featq   = (_Float16*)take((size_t)BATCH*MS*256*2);
  _Float16* featk   = (_Float16*)take((size_t)BATCH*MS*16*256*2);
  _Float16* q16     = (_Float16*)take((size_t)BATCH*MS*256*2);
  float*    kt      = (float*)   take((size_t)BATCH*MS*256*4);

  float* downx = (float*)d_out;                       // (B,M,3)
  float* newx  = (float*)d_out + (size_t)BATCH*MS*3;  // (B,M,3)

  // 1. weight prep (f16 copies + k_w transpose)
  prep_kernel<<<256, 256, 0, stream>>>(csw1, csw2, qw, kw, csw1_16, csw2_16, qw16, kwT16);
  // 2. global feature MLP
  gfeat_kernel<<<BATCH, 256, 0, stream>>>(gfeat, cfw1, cfb1, cfw2, cfb2, g16);
  // 3. per-point MLP + concat -> xcat f16 [B*N][256]
  pointmlp_kernel<<<(BATCH*NPTS)/256, 256, 0, stream>>>(x, c1w1, c1b1, c1w2, c1b2, g16, xcat);
  // 4. cs MLP via WMMA: 256->256 relu -> hbuf; 256->128 -> fbuf
  gemm_wmma<1,1><<<(16*((BATCH*NPTS)/16) + 7)/8, 256, 0, stream>>>(
      csw1_16, xcat, csb1, hbuf, nullptr, 256, 256, BATCH*NPTS);
  gemm_wmma<0,1><<<(8*((BATCH*NPTS)/16) + 7)/8, 256, 0, stream>>>(
      csw2_16, hbuf, csb2, fbuf, nullptr, 128, 256, BATCH*NPTS);
  // 5. FPS (emits down_x into d_out)
  fps_kernel<<<BATCH, 256, 0, stream>>>(x, fpsidx, downx);
  // 6. KNN (17 nearest, self first)
  knn_kernel<<<BATCH*MS, 256, 0, stream>>>(x, downx, knnidx);
  // 7. local geometry convs + feature assembly
  local_kernel<<<BATCH*MS, 128, 0, stream>>>(x, fbuf, knnidx, downx,
      s2w1, s2b1, s2g1, s2bb1, s2w2, s2b2, s2g2, s2bb2, s2w3, s2b3, featq, featk);
  // 8. q = Q_w * featq + q_b  (WMMA, f16 out)
  gemm_wmma<0,1><<<(16*((BATCH*MS)/16) + 7)/8, 256, 0, stream>>>(
      qw16, featq, qb, q16, nullptr, 256, 256, BATCH*MS);
  // 9. kt = K_w^T * q  (WMMA, f32 out; k_b handled implicitly — softmax invariant)
  gemm_wmma<0,0><<<(16*((BATCH*MS)/16) + 7)/8, 256, 0, stream>>>(
      kwT16, q16, nullptr, nullptr, kt, 256, 256, BATCH*MS);
  // 10. attention softmax + weighted neighbor coords -> new_x
  attn_kernel<<<BATCH*MS, 32, 0, stream>>>(kt, featk, knnidx, x, newx);
}
